// Top2MoE_84164179132471
// MI455X (gfx1250) — matrix operand order repaired and checked by reading, not compiled, not run on hardware
//
#include <hip/hip_runtime.h>
#include <cstdint>
#include <cstddef>

typedef __bf16 bf16;
typedef bf16 bf16x16 __attribute__((ext_vector_type(16)));
typedef float f32x8 __attribute__((ext_vector_type(8)));

union FragAB { bf16x16 v; uint4 q[2]; };

#define BLK_M 256
#define BLK_N 128
#define BLK_K 32
#define LDT   40   // LDS row stride in bf16 elems (80B: 16B-aligned, breaks 64-bank conflicts)

#if __has_builtin(__builtin_amdgcn_global_load_async_to_lds_b128) && \
    __has_builtin(__builtin_amdgcn_s_wait_asynccnt)
#define USE_ASYNC_LDS 1
#else
#define USE_ASYNC_LDS 0
#endif

#if USE_ASYNC_LDS
typedef int v4i __attribute__((vector_size(16)));
typedef __attribute__((address_space(1))) v4i gv4i;   // global int4
typedef __attribute__((address_space(3))) v4i lv4i;   // LDS int4
__device__ __forceinline__ void async_cp16(const bf16* g, bf16* l) {
  // GLOBAL_LOAD_ASYNC_TO_LDS_B128: 16B per lane, tracked by ASYNCcnt
  __builtin_amdgcn_global_load_async_to_lds_b128((gv4i*)g, (lv4i*)l, 0, 0);
}
#endif

// ---------------- weight convert: f32 -> bf16 ----------------
__global__ __launch_bounds__(256) void cvt_w_kernel(
    const float* __restrict__ w1, const float* __restrict__ w2,
    bf16* __restrict__ o1, bf16* __restrict__ o2, size_t n) {
  size_t stride = (size_t)gridDim.x * blockDim.x;
  for (size_t i = (size_t)blockIdx.x * blockDim.x + threadIdx.x; i < n; i += stride) {
    o1[i] = (bf16)w1[i];
    o2[i] = (bf16)w2[i];
  }
}

// ---------------- gating: one wave per token ----------------
__global__ __launch_bounds__(256) void gate_kernel(
    const float* __restrict__ x, const float* __restrict__ gw,
    int N, int H,
    int* __restrict__ eid, float* __restrict__ ew, int* __restrict__ counts) {
  int wid  = (int)((blockIdx.x * (size_t)blockDim.x + threadIdx.x) >> 5);
  int lane = threadIdx.x & 31;
  if (wid >= N) return;
  const float* xr = x + (size_t)wid * H;
  float acc[8];
#pragma unroll
  for (int e = 0; e < 8; ++e) acc[e] = 0.0f;
  for (int k = lane; k < H; k += 32) {
    float xv = xr[k];
#pragma unroll
    for (int e = 0; e < 8; ++e) acc[e] += xv * gw[e * H + k];
  }
#pragma unroll
  for (int e = 0; e < 8; ++e) {
#pragma unroll
    for (int m = 16; m > 0; m >>= 1) acc[e] += __shfl_xor(acc[e], m, 32);
  }
  if (lane == 0) {
    float mx = acc[0];
#pragma unroll
    for (int e = 1; e < 8; ++e) mx = fmaxf(mx, acc[e]);
    float p[8], s = 0.0f;
#pragma unroll
    for (int e = 0; e < 8; ++e) { p[e] = __expf(acc[e] - mx); s += p[e]; }
    float inv = 1.0f / s;
#pragma unroll
    for (int e = 0; e < 8; ++e) p[e] *= inv;
    int e0 = 0;
#pragma unroll
    for (int e = 1; e < 8; ++e) if (p[e] > p[e0]) e0 = e;
    int e1 = (e0 == 0) ? 1 : 0;
#pragma unroll
    for (int e = 0; e < 8; ++e) if (e != e0 && p[e] > p[e1]) e1 = e;
    eid[2 * wid] = e0;  eid[2 * wid + 1] = e1;
    ew[2 * wid] = p[e0]; ew[2 * wid + 1] = p[e1];
    atomicAdd(&counts[e0], 1);
    atomicAdd(&counts[e1], 1);
  }
}

// ---------------- keep flags ----------------
__global__ void keep_kernel(const int* __restrict__ counts, int* __restrict__ keep,
                            int cap, int E) {
  int e = threadIdx.x;
  if (e < E) keep[e] = (counts[e] <= cap) ? 1 : 0;
}

// ---------------- per-assignment slot via atomics ----------------
__global__ __launch_bounds__(256) void pos_kernel(
    const int* __restrict__ eid, const int* __restrict__ keep,
    int* __restrict__ posc, int* __restrict__ pp, int nA, int cap) {
  int a = blockIdx.x * blockDim.x + threadIdx.x;
  if (a >= nA) return;
  int e = eid[a];
  pp[a] = keep[e] ? atomicAdd(&posc[e], 1) : cap;
}

// ---------------- dispatch tokens into capacity buffer (f32 -> bf16) ----------------
__global__ __launch_bounds__(256) void dispatch_kernel(
    const float* __restrict__ x, const int* __restrict__ eid,
    const int* __restrict__ pp, bf16* __restrict__ buf, int H, int cap) {
  int a = blockIdx.x;
  int e = eid[a], p = pp[a], tok = a >> 1;
  const float* xr = x + (size_t)tok * H;
  bf16* dst = buf + ((size_t)e * (cap + 1) + p) * (size_t)H;
  for (int k = threadIdx.x; k < H; k += 256) dst[k] = (bf16)xr[k];
}

// ---------------- grouped expert GEMM: D = A * W^T + b, optional SiLU ----------------
// A: [E][M][K] bf16, W: [E][Nn][K] bf16 (row-major => already "B transposed" for WMMA).
// Block tile 256x128, 8 waves as 4(M)x2(N), each wave 64x64 = 16 WMMAs per k-step
// (16 ds_load_b128 : 16 wmma). Double-buffered LDS fed by async-to-LDS copies.
// Only rows < counts[e] (kept experts) are ever gathered downstream, so blocks whose
// M-tile lies entirely in unused rows exit early; row clamp keeps async issue uniform.
// ACT==0: out = silu(..) as bf16     ACT==1: out = .. as f32
template <int ACT>
__global__ __launch_bounds__(256) void gemm_kernel(
    const bf16* __restrict__ A, const bf16* __restrict__ W,
    const float* __restrict__ bias,
    const int* __restrict__ counts, const int* __restrict__ keep,
    bf16* __restrict__ outB, float* __restrict__ outF,
    int M, int K, int Nn) {
  __shared__ bf16 sA[2][BLK_M * LDT];   // 2 x 20KB
  __shared__ bf16 sB[2][BLK_N * LDT];   // 2 x 10KB

  const int e  = blockIdx.z;
  const int rowsUsed = keep[e] ? counts[e] : 0;
  const int m0 = blockIdx.y * BLK_M;
  if (m0 >= rowsUsed) return;               // uniform early-exit: tile never consumed
  const int Meff = (rowsUsed < M) ? rowsUsed : M;

  const int n0  = blockIdx.x * BLK_N;
  const int tid = threadIdx.x;
  const int lane = tid & 31;
  const int wid  = tid >> 5;
  const int wm = (wid & 3) * 64;   // 4 wave rows of 64
  const int wn = (wid >> 2) * 64;  // 2 wave cols of 64
  const int hs = lane >> 4;        // half-wave select
  const int lr = lane & 15;

  const bf16* Ae  = A + (size_t)e * M * K;
  const bf16* We  = W + (size_t)e * Nn * K;
  const float* be = bias + (size_t)e * Nn;

  // branchless tile issue: A = 1024 chunks (4/thread), B = 512 chunks (2/thread)
  auto issue_tile = [&](int k0, int db) {
#pragma unroll
    for (int h = 0; h < 4; ++h) {
      int idx = tid + h * 256;        // 0..1023
      int row = idx >> 2;             // 0..255
      int kc  = (idx & 3) * 8;        // 0,8,16,24
      int gr  = m0 + row;
      gr = (gr < Meff) ? gr : (Meff - 1);   // clamp: garbage rows never stored
#if USE_ASYNC_LDS
      async_cp16(Ae + (size_t)gr * K + k0 + kc, &sA[db][row * LDT + kc]);
#else
      *(uint4*)&sA[db][row * LDT + kc] = *(const uint4*)(Ae + (size_t)gr * K + k0 + kc);
#endif
    }
#pragma unroll
    for (int h = 0; h < 2; ++h) {
      int idx = tid + h * 256;        // 0..511
      int row = idx >> 2;             // 0..127
      int kc  = (idx & 3) * 8;
#if USE_ASYNC_LDS
      async_cp16(We + (size_t)(n0 + row) * K + k0 + kc, &sB[db][row * LDT + kc]);
#else
      *(uint4*)&sB[db][row * LDT + kc] = *(const uint4*)(We + (size_t)(n0 + row) * K + k0 + kc);
#endif
    }
  };

  f32x8 acc[4][4];
#pragma unroll
  for (int i = 0; i < 4; ++i)
#pragma unroll
    for (int j = 0; j < 4; ++j)
#pragma unroll
      for (int r = 0; r < 8; ++r) acc[i][j][r] = 0.0f;

  const int kTiles = K / BLK_K;
  issue_tile(0, 0);
  for (int kt = 0; kt < kTiles; ++kt) {
    const int cur = kt & 1;
    if (kt + 1 < kTiles) {
      issue_tile((kt + 1) * BLK_K, cur ^ 1);   // prefetch next tile into other buffer
#if USE_ASYNC_LDS
      __builtin_amdgcn_s_wait_asynccnt(6);     // own 6 current-tile copies done (in-order)
#endif
    } else {
#if USE_ASYNC_LDS
      __builtin_amdgcn_s_wait_asynccnt(0);
#endif
    }
    __syncthreads();                           // publish current tile across waves

    FragAB a[4], b[4];
#pragma unroll
    for (int i = 0; i < 4; ++i) {  // A 16x32 frag: lanes 0-15 K=0..7 & 16..23, lanes 16-31 K=8..15 & 24..31
      int r = wm + i * 16 + lr;
      a[i].q[0] = *(const uint4*)&sA[cur][r * LDT + hs * 8];
      a[i].q[1] = *(const uint4*)&sA[cur][r * LDT + hs * 8 + 16];
    }
#pragma unroll
    for (int j = 0; j < 4; ++j) {  // B 32x16 frag: lanes 0-15 K=0..7 & 16..23, lanes 16-31 K=8..15 & 24..31 (col = lane%16)
      int r = wn + j * 16 + lr;
      b[j].q[0] = *(const uint4*)&sB[cur][r * LDT + hs * 8];
      b[j].q[1] = *(const uint4*)&sB[cur][r * LDT + hs * 8 + 16];
    }
#pragma unroll
    for (int i = 0; i < 4; ++i)
#pragma unroll
      for (int j = 0; j < 4; ++j)
        acc[i][j] = __builtin_amdgcn_wmma_f32_16x16x32_bf16(
            false, a[i].v, false, b[j].v, (short)0, acc[i][j], false, false);
    __syncthreads();                           // reads done before buffer reuse
  }

  // epilogue: D layout — VGPR r, lanes 0-15 row=r, lanes 16-31 row=8+r, col=lane%16
#pragma unroll
  for (int i = 0; i < 4; ++i) {
    int rb = m0 + wm + i * 16 + hs * 8;
#pragma unroll
    for (int j = 0; j < 4; ++j) {
      int cb = n0 + wn + j * 16 + lr;
      float bv = be[cb];
#pragma unroll
      for (int r = 0; r < 8; ++r) {
        int grow = rb + r;
        if (grow < Meff) {
          float v = acc[i][j][r] + bv;
          if (ACT == 0) {
            float s = v / (1.0f + __expf(-v));  // silu
            outB[(size_t)e * M * Nn + (size_t)grow * Nn + cb] = (bf16)s;
          } else {
            outF[(size_t)e * M * Nn + (size_t)grow * Nn + cb] = v;
          }
        }
      }
    }
  }
}

// ---------------- combine: out = w0*valid0*y[e0,p0] + w1*valid1*y[e1,p1] ----------------
// Zero-weight terms are skipped entirely so rows that were never computed are never read.
__global__ __launch_bounds__(256) void combine_kernel(
    const float* __restrict__ y, const int* __restrict__ eid,
    const float* __restrict__ ew, const int* __restrict__ pp,
    const int* __restrict__ keep, float* __restrict__ out, int H, int cap) {
  int tok = blockIdx.x;
  int e0 = eid[2 * tok], e1 = eid[2 * tok + 1];
  int p0 = pp[2 * tok],  p1 = pp[2 * tok + 1];
  float w0 = ew[2 * tok]     * (keep[e0] ? 1.0f : 0.0f);
  float w1 = ew[2 * tok + 1] * (keep[e1] ? 1.0f : 0.0f);
  const float* y0 = y + ((size_t)e0 * (cap + 1) + p0) * (size_t)H;
  const float* y1 = y + ((size_t)e1 * (cap + 1) + p1) * (size_t)H;
  float* o = out + (size_t)tok * H;
  for (int k = threadIdx.x; k < H; k += 256) {
    float v = 0.0f;
    if (w0 != 0.0f) v += w0 * y0[k];
    if (w1 != 0.0f) v += w1 * y1[k];
    o[k] = v;
  }
}

extern "C" void kernel_launch(void* const* d_in, const int* in_sizes, int n_in,
                              void* d_out, int out_size, void* d_ws, size_t ws_size,
                              hipStream_t stream) {
  const float* tokens = (const float*)d_in[0];
  const float* gate_w = (const float*)d_in[1];
  const float* w1     = (const float*)d_in[2];
  const float* b1     = (const float*)d_in[3];
  const float* w2     = (const float*)d_in[4];
  const float* b2     = (const float*)d_in[5];
  float* out = (float*)d_out;

  const long long EH  = in_sizes[1];
  const long long EHH = in_sizes[2];
  const int H = (int)(EHH / EH);
  const int E = (int)(EH / H);
  const int N = (int)(in_sizes[0] / H);
  const int cap = (int)((5LL * N) / (4LL * E));  // floor(1.25*N/E)
  const int Mrows = cap + 1;
  const int nA = 2 * N;

  uint8_t* ws = (uint8_t*)d_ws;
  size_t off = 0;
  auto alloc = [&](size_t bytes) {
    size_t o = off;
    off += (bytes + 255) & ~(size_t)255;
    return o;
  };
  int*   counts = (int*)(ws + alloc((size_t)E * 4));
  int*   posc   = (int*)(ws + alloc((size_t)E * 4));
  int*   keep   = (int*)(ws + alloc((size_t)E * 4));
  int*   eid    = (int*)(ws + alloc((size_t)nA * 4));
  float* ew     = (float*)(ws + alloc((size_t)nA * 4));
  int*   pp     = (int*)(ws + alloc((size_t)nA * 4));
  bf16*  w1b    = (bf16*)(ws + alloc((size_t)EHH * 2));
  bf16*  w2b    = (bf16*)(ws + alloc((size_t)EHH * 2));
  const size_t bufElems = (size_t)E * Mrows * H;
  bf16*  buf    = (bf16*)(ws + alloc(bufElems * 2));
  bf16*  hb     = (bf16*)(ws + alloc(bufElems * 2));
  float* yb     = (float*)(ws + alloc(bufElems * 4));
  (void)ws_size; (void)n_in; (void)out_size;

  // zero counters only (counts/posc/keep in the first 3 * 256B slots); the big
  // capacity buffer needs no memset: every consumed row is rewritten each call.
  (void)hipMemsetAsync(d_ws, 0, 768, stream);

  cvt_w_kernel<<<4096, 256, 0, stream>>>(w1, w2, w1b, w2b, (size_t)EHH);

  gate_kernel<<<(N * 32 + 255) / 256, 256, 0, stream>>>(tokens, gate_w, N, H,
                                                        eid, ew, counts);
  keep_kernel<<<1, 32, 0, stream>>>(counts, keep, cap, E);
  pos_kernel<<<(nA + 255) / 256, 256, 0, stream>>>(eid, keep, posc, pp, nA, cap);
  dispatch_kernel<<<nA, 256, 0, stream>>>(tokens, eid, pp, buf, H, cap);

  dim3 gg(H / BLK_N, (Mrows + BLK_M - 1) / BLK_M, E);
  gemm_kernel<0><<<gg, 256, 0, stream>>>(buf, w1b, b1, counts, keep, hb, nullptr,
                                         Mrows, H, H);
  gemm_kernel<1><<<gg, 256, 0, stream>>>(hb, w2b, b2, counts, keep, nullptr, yb,
                                         Mrows, H, H);

  combine_kernel<<<N, 256, 0, stream>>>(yb, eid, ew, pp, keep, out, H, cap);
}
